// TrackMPNN_29472065585913
// MI455X (gfx1250) — matrix operand order repaired and checked by reading, not compiled, not run on hardware
//
#include <hip/hip_runtime.h>
#include <hip/hip_bf16.h>

typedef __attribute__((ext_vector_type(16))) _Float16 v16h;
typedef __attribute__((ext_vector_type(8)))  _Float16 v8h;
typedef __attribute__((ext_vector_type(4)))  _Float16 v4h;
typedef __attribute__((ext_vector_type(8)))  float    v8f;

#define NTOT 4096
#define NNEW 2048
#define HID  64
#define LEAK 0.2f
#define NEGINF (-1.0e9f)
#define BNEPS 1e-5f

// ---------------------------------------------------------------------------
// WMMA GEMM: C[M x Ntot] = A[M x K](f16 row-major) * Bp(packed f16).
// One wave computes a 16x64 slab (4 column tiles) so A is loaded once per
// k-step and feeds 4 v_wmma ops.  grid = (M/128, Ntot/64), block = 256.
// Bp layout: per (kb = K/32, nb = N/16) a 32-lane x 16-half record:
//   lane<16  -> column nb*16+lane,    rows k0..k0+15
//   lane>=16 -> column nb*16+lane-16, rows k0+16..k0+31
// A per-lane loads match the Bp k order: lanes 0-15 take K offsets
// {0..15}, lanes 16-31 take {16..31}.
// ---------------------------------------------------------------------------
__global__ __launch_bounds__(256) void k_wmma_gemm(
    const _Float16* __restrict__ A, int lda,
    const _Float16* __restrict__ Bp, int K, int Ntot,
    const float* __restrict__ bias,
    float* __restrict__ Cf, int ldc,
    _Float16* __restrict__ Ch, int ldch)
{
  const int lane = threadIdx.x & 31;
  const int wave = threadIdx.x >> 5;
  const int m0  = (blockIdx.x * 8 + wave) * 16;
  const int tnb = blockIdx.y * 4;           // first of 4 consecutive col tiles
  const int n16 = Ntot >> 4;
  const int rowA = m0 + (lane & 15);
  const int ksel = (lane & 16) ? 16 : 0;
  v8f acc[4] = {{}, {}, {}, {}};
  for (int k0 = 0; k0 < K; k0 += 32) {
    const _Float16* ap = A + (size_t)rowA * lda + k0 + ksel;
    v8h alo = *(const v8h*)(ap);
    v8h ahi = *(const v8h*)(ap + 8);
    v16h av = __builtin_shufflevector(alo, ahi,
        0,1,2,3,4,5,6,7,8,9,10,11,12,13,14,15);
    const _Float16* bp = Bp + ((size_t)(k0 >> 5) * n16 + tnb) * 512 + lane * 16;
#pragma unroll
    for (int nt = 0; nt < 4; ++nt) {
      v16h bv = *(const v16h*)(bp + nt * 512);
      acc[nt] = __builtin_amdgcn_wmma_f32_16x16x32_f16(
          false, av, false, bv, (short)0, acc[nt], false, false);
    }
  }
  const int rb = m0 + ((lane & 16) ? 8 : 0);
#pragma unroll
  for (int nt = 0; nt < 4; ++nt) {
    const int col = (tnb + nt) * 16 + (lane & 15);
    const float bb = bias ? bias[col] : 0.0f;
#pragma unroll
    for (int r = 0; r < 8; ++r) {
      float v = acc[nt][r] + bb;
      if (Cf) Cf[(size_t)(rb + r) * ldc  + col] = v;
      if (Ch) Ch[(size_t)(rb + r) * ldch + col] = (_Float16)v;
    }
  }
}

// Pack f32 [K x ncols] row-major weight into WMMA B-tile f16 layout
__global__ __launch_bounds__(256) void k_pack_b(
    const float* __restrict__ B, int ldb, int K, int ncols,
    _Float16* __restrict__ Bp, int Ntot, int colOff)
{
  int idx = blockIdx.x * blockDim.x + threadIdx.x;
  if (idx >= K * ncols) return;
  int k = idx / ncols, c0 = idx - k * ncols;
  int c = c0 + colOff;
  int kb = k >> 5, krel = k & 31;
  int lane = (krel < 16) ? (c & 15) : (16 + (c & 15));
  size_t off = ((size_t)kb * (Ntot >> 4) + (c >> 4)) * 512 + lane * 16 + (krel & 15);
  Bp[off] = (_Float16)B[(size_t)k * ldb + c0];
}

// MLP layer 1: h1 = x[:, xoff:xoff+nin] @ W1 + b1     [NNEW x HID]
__global__ __launch_bounds__(256) void k_mlp1(
    const float* __restrict__ x, int nin, int xoff,
    const float* __restrict__ W1, const float* __restrict__ b1,
    float* __restrict__ h1)
{
  int idx = blockIdx.x * blockDim.x + threadIdx.x;
  if (idx >= NNEW * HID) return;
  int r = idx >> 6, c = idx & 63;
  float s = b1[c];
  for (int k = 0; k < nin; ++k) s += x[r * 10 + xoff + k] * W1[k * HID + c];
  h1[idx] = s;
}

// Deterministic per-column batch stats: stats[c]=mean, stats[HID+c]=biased var
__global__ __launch_bounds__(256) void k_colstats(
    const float* __restrict__ h1, float* __restrict__ stats)
{
  const int c = blockIdx.x;
  float s = 0.f, s2 = 0.f;
  for (int r = threadIdx.x; r < NNEW; r += 256) {
    float v = h1[(size_t)r * HID + c]; s += v; s2 += v * v;
  }
  __shared__ float sh[256], sh2[256];
  sh[threadIdx.x] = s; sh2[threadIdx.x] = s2; __syncthreads();
  for (int o = 128; o; o >>= 1) {
    if (threadIdx.x < o) { sh[threadIdx.x] += sh[threadIdx.x + o];
                           sh2[threadIdx.x] += sh2[threadIdx.x + o]; }
    __syncthreads();
  }
  if (threadIdx.x == 0) {
    float mu = sh[0] / (float)NNEW;
    stats[c] = mu;
    stats[HID + c] = sh2[0] / (float)NNEW - mu * mu;
  }
}

// BN + ReLU -> f16 A-matrix for the W2 GEMM
__global__ __launch_bounds__(256) void k_mlp2(
    const float* __restrict__ h1, const float* __restrict__ stats,
    const float* __restrict__ gamma, const float* __restrict__ beta,
    _Float16* __restrict__ a16)
{
  int idx = blockIdx.x * blockDim.x + threadIdx.x;
  if (idx >= NNEW * HID) return;
  int c = idx & 63;
  float mu = stats[c], var = stats[HID + c];
  float v = (h1[idx] - mu) * rsqrtf(var + BNEPS) * gamma[c] + beta[c];
  a16[idx] = (_Float16)fmaxf(v, 0.f);
}

// Build h = concat(h_in_slice, d_node * xs): f32, f16 row-major, and packed-B
__global__ __launch_bounds__(256) void k_build_h(
    const float* __restrict__ h_in, int s, const float* __restrict__ xs,
    const float* __restrict__ nodeA,
    float* __restrict__ hF, _Float16* __restrict__ hRM, _Float16* __restrict__ hPK)
{
  int idx = blockIdx.x * blockDim.x + threadIdx.x;
  if (idx >= NTOT * HID) return;
  int r = idx >> 6, c = idx & 63;
  float v;
  if (r < NNEW) v = h_in[(size_t)r * (2 * HID) + s * HID + c];
  else v = xs[(size_t)(r - NNEW) * HID + c] * nodeA[(size_t)r * NTOT + r];
  hF[idx] = v;
  hRM[idx] = (_Float16)v;
  int kb = r >> 5, krel = r & 31;
  int lane = (krel < 16) ? (c & 15) : (16 + (c & 15));
  size_t off = ((size_t)kb * (HID >> 4) + (c >> 4)) * 512 + lane * 16 + (krel & 15);
  hPK[off] = (_Float16)v;
}

// src = h @ att_src, dst = h @ att_dst   (NHEADS = 1)
__global__ __launch_bounds__(256) void k_srcdst(
    const float* __restrict__ hF, const float* __restrict__ asrc,
    const float* __restrict__ adst, float* __restrict__ srcv, float* __restrict__ dstv)
{
  int i = blockIdx.x * blockDim.x + threadIdx.x;
  if (i >= NTOT) return;
  float s1 = 0.f, s2 = 0.f;
  for (int d = 0; d < HID; ++d) {
    float v = hF[(size_t)i * HID + d];
    s1 += v * asrc[d]; s2 += v * adst[d];
  }
  srcv[i] = s1; dstv[i] = s2;
}

// Masked leaky-relu scores + row softmax. One block per row, scores in LDS.
// All global traffic is b128 (float4) / b64 (4 x f16) vectorized.
__global__ __launch_bounds__(256) void k_attn_row(
    const float* __restrict__ nodeA, const float* __restrict__ edgeA,
    const float* __restrict__ srcv, const float* __restrict__ dstv,
    float* __restrict__ alphaF, _Float16* __restrict__ alphaH)
{
  __shared__ float sc[NTOT];
  __shared__ float red[8];
  __shared__ float bval;
  const int i = blockIdx.x;
  const int tid = threadIdx.x, lane = tid & 31, wave = tid >> 5;
  const float si = srcv[i];
  const float4* nrow = (const float4*)(nodeA + (size_t)i * NTOT);
  const float4* erow = (const float4*)(edgeA + (size_t)i * NTOT);
  const float4* dv4  = (const float4*)dstv;
  float lmax = NEGINF;
  for (int q = tid; q < NTOT / 4; q += 256) {
    if (i + 1 < NTOT) {
      __builtin_prefetch(&nodeA[(size_t)(i + 1) * NTOT + q * 4], 0, 1);
      __builtin_prefetch(&edgeA[(size_t)(i + 1) * NTOT + q * 4], 0, 1);
    }
    float4 na = nrow[q], ea = erow[q], dd = dv4[q];
    float a[4] = { na.x + ea.x, na.y + ea.y, na.z + ea.z, na.w + ea.w };
    float d[4] = { dd.x, dd.y, dd.z, dd.w };
#pragma unroll
    for (int j = 0; j < 4; ++j) {
      float sv;
      if (a[j] != 0.f) { float t = si + d[j]; sv = (t > 0.f) ? t : LEAK * t; }
      else sv = NEGINF;
      sc[q * 4 + j] = sv;
      lmax = fmaxf(lmax, sv);
    }
  }
  for (int o = 16; o; o >>= 1) lmax = fmaxf(lmax, __shfl_xor(lmax, o));
  if (lane == 0) red[wave] = lmax;
  __syncthreads();
  if (tid == 0) { float m = red[0]; for (int wj = 1; wj < 8; ++wj) m = fmaxf(m, red[wj]); bval = m; }
  __syncthreads();
  const float rmax = bval;
  float lsum = 0.f;
  for (int c = tid; c < NTOT; c += 256) {
    float sv = sc[c];
    float e = (sv > -1.0e8f) ? __expf(sv - rmax) : 0.f;
    sc[c] = e; lsum += e;
  }
  for (int o = 16; o; o >>= 1) lsum += __shfl_xor(lsum, o);
  if (lane == 0) red[wave] = lsum;
  __syncthreads();
  if (tid == 0) { float t = 0.f; for (int wj = 0; wj < 8; ++wj) t += red[wj]; bval = 1.f / t; }
  __syncthreads();
  const float inv = bval;
  float4* aF4 = (float4*)(alphaF + (size_t)i * NTOT);
  v4h*    aH4 = (v4h*)(alphaH + (size_t)i * NTOT);
  for (int q = tid; q < NTOT / 4; q += 256) {
    float4 av;
    av.x = sc[q * 4 + 0] * inv;           // masked entries are exactly 0
    av.y = sc[q * 4 + 1] * inv;
    av.z = sc[q * 4 + 2] * inv;
    av.w = sc[q * 4 + 3] * inv;
    aF4[q] = av;
    v4h hv; hv[0] = (_Float16)av.x; hv[1] = (_Float16)av.y;
    hv[2] = (_Float16)av.z; hv[3] = (_Float16)av.w;
    aH4[q] = hv;
  }
}

// mcat = [h | m_agg] as f16 row-major [NTOT x 128]
__global__ __launch_bounds__(256) void k_build_mcat(
    const float* __restrict__ hF, const float* __restrict__ mAgg,
    _Float16* __restrict__ mcat)
{
  int idx = blockIdx.x * blockDim.x + threadIdx.x;
  if (idx >= NTOT * 128) return;
  int r = idx >> 7, c = idx & 127;
  float v = (c < HID) ? hF[(size_t)r * HID + c] : mAgg[(size_t)r * HID + (c - HID)];
  mcat[idx] = (_Float16)v;
}

// GRU gate fusion: GM = m@[Wz|Wr|Wn], GH = h@[Uz|Ur|Un]  (both [NTOT x 192])
__global__ __launch_bounds__(256) void k_gru(
    const float* __restrict__ GMp, const float* __restrict__ GHp,
    const float* __restrict__ bz, const float* __restrict__ br,
    const float* __restrict__ bn, const float* __restrict__ hF,
    int s, float* __restrict__ hout)
{
  int idx = blockIdx.x * blockDim.x + threadIdx.x;
  if (idx >= NTOT * HID) return;
  int r = idx >> 6, c = idx & 63;
  const float* gm = GMp + (size_t)r * 192;
  const float* gh = GHp + (size_t)r * 192;
  float z  = 1.f / (1.f + __expf(-(gm[c]        + gh[c]        + bz[c])));
  float rr = 1.f / (1.f + __expf(-(gm[64 + c]   + gh[64 + c]   + br[c])));
  float n  = tanhf(gm[128 + c] + rr * gh[128 + c] + bn[c]);
  float h  = hF[idx];
  hout[(size_t)r * 128 + s * HID + c] = (1.f - z) * n + z * h;
}

// y = d_node*(h_out@Wn_out+bn) + d_edge*(h_out@We_out+be); sigmoid
__global__ __launch_bounds__(256) void k_final(
    const float* __restrict__ hout,
    const float* __restrict__ Wn, const float* __restrict__ bn,
    const float* __restrict__ We, const float* __restrict__ be,
    const float* __restrict__ nodeA, const float* __restrict__ edgeA,
    float* __restrict__ outSig, float* __restrict__ outY)
{
  int r = blockIdx.x * blockDim.x + threadIdx.x;
  if (r >= NTOT) return;
  float an = bn[0], ae = be[0];
  const float* ho = hout + (size_t)r * 128;
  for (int d = 0; d < 128; ++d) { float v = ho[d]; an += v * Wn[d]; ae += v * We[d]; }
  float dn = nodeA[(size_t)r * NTOT + r], de = edgeA[(size_t)r * NTOT + r];
  float y = dn * an + de * ae;
  outY[r] = y;
  outSig[r] = 1.f / (1.f + __expf(-y));
}

extern "C" void kernel_launch(void* const* d_in, const int* in_sizes, int n_in,
                              void* d_out, int out_size, void* d_ws, size_t ws_size,
                              hipStream_t stream)
{
  (void)in_sizes; (void)n_in; (void)out_size; (void)ws_size;
  const float* x     = (const float*)d_in[0];
  const float* h_in  = (const float*)d_in[1];
  const float* nodeA = (const float*)d_in[2];
  const float* edgeA = (const float*)d_in[3];
  // params flattened in insertion order:
  // in{s}: W1,b1,gamma,beta,W2,b2   gru{s}: att_src,att_dst,Wm,bm,Wz,Uz,bz,Wr,Ur,br,Wn,Un,bn
  auto INP = [&](int s, int k) { return (const float*)d_in[4 + s * 19 + k]; };
  auto GRU = [&](int s, int k) { return (const float*)d_in[10 + s * 19 + k]; };
  const float* WnO = (const float*)d_in[42];
  const float* bnO = (const float*)d_in[43];
  const float* WeO = (const float*)d_in[44];
  const float* beO = (const float*)d_in[45];

  float* out    = (float*)d_out;
  float* outSig = out;
  float* outY   = out + NTOT;
  float* outH   = out + 2 * NTOT;                      // [NTOT x 128]
  float* outA0  = outH + (size_t)NTOT * 128;           // [NTOT x NTOT]
  float* outA1  = outA0 + (size_t)NTOT * NTOT;

  char* w = (char*)d_ws;
  auto alloc = [&](size_t b) -> void* {
    void* p = (void*)w; w += (b + 255) & ~(size_t)255; return p;
  };

  for (int s = 0; s < 2; ++s) {
    const int nin = s ? 2 : 8, xoff = s ? 8 : 0;
    float*    h1    = (float*)   alloc((size_t)NNEW * HID * 4);
    float*    stats = (float*)   alloc(2 * HID * 4);
    _Float16* a16   = (_Float16*)alloc((size_t)NNEW * HID * 2);
    float*    xs    = (float*)   alloc((size_t)NNEW * HID * 4);
    float*    hF    = (float*)   alloc((size_t)NTOT * HID * 4);
    _Float16* hRM   = (_Float16*)alloc((size_t)NTOT * HID * 2);
    _Float16* hPK   = (_Float16*)alloc((size_t)NTOT * HID * 2);
    _Float16* al16  = (_Float16*)alloc((size_t)NTOT * NTOT * 2);
    float*    mAgg  = (float*)   alloc((size_t)NTOT * HID * 4);
    _Float16* mcat  = (_Float16*)alloc((size_t)NTOT * 128 * 2);
    _Float16* m16   = (_Float16*)alloc((size_t)NTOT * HID * 2);
    float*    GMb   = (float*)   alloc((size_t)NTOT * 192 * 4);
    float*    GHb   = (float*)   alloc((size_t)NTOT * 192 * 4);
    _Float16* W2p   = (_Float16*)alloc(64 * 64 * 2);
    _Float16* Wmp   = (_Float16*)alloc(128 * 64 * 2);
    _Float16* Wzp   = (_Float16*)alloc(64 * 192 * 2);
    _Float16* Uzp   = (_Float16*)alloc(64 * 192 * 2);
    float*    srcv  = (float*)   alloc(NTOT * 4);
    float*    dstv  = (float*)   alloc(NTOT * 4);
    float*    alphaF = s ? outA1 : outA0;

    // ---- weight packing into WMMA B layout ----
    k_pack_b<<<(64 * 64 + 255) / 256, 256, 0, stream>>>(INP(s, 4), 64, 64, 64, W2p, 64, 0);
    k_pack_b<<<(128 * 64 + 255) / 256, 256, 0, stream>>>(GRU(s, 2), 64, 128, 64, Wmp, 64, 0);
    k_pack_b<<<(64 * 64 + 255) / 256, 256, 0, stream>>>(GRU(s, 4),  64, 64, 64, Wzp, 192, 0);
    k_pack_b<<<(64 * 64 + 255) / 256, 256, 0, stream>>>(GRU(s, 7),  64, 64, 64, Wzp, 192, 64);
    k_pack_b<<<(64 * 64 + 255) / 256, 256, 0, stream>>>(GRU(s, 10), 64, 64, 64, Wzp, 192, 128);
    k_pack_b<<<(64 * 64 + 255) / 256, 256, 0, stream>>>(GRU(s, 5),  64, 64, 64, Uzp, 192, 0);
    k_pack_b<<<(64 * 64 + 255) / 256, 256, 0, stream>>>(GRU(s, 8),  64, 64, 64, Uzp, 192, 64);
    k_pack_b<<<(64 * 64 + 255) / 256, 256, 0, stream>>>(GRU(s, 11), 64, 64, 64, Uzp, 192, 128);

    // ---- MLP (Linear -> BN(batch stats) -> ReLU -> Linear via WMMA) ----
    k_mlp1<<<(NNEW * HID + 255) / 256, 256, 0, stream>>>(x, nin, xoff, INP(s, 0), INP(s, 1), h1);
    k_colstats<<<HID, 256, 0, stream>>>(h1, stats);
    k_mlp2<<<(NNEW * HID + 255) / 256, 256, 0, stream>>>(h1, stats, INP(s, 2), INP(s, 3), a16);
    k_wmma_gemm<<<dim3(NNEW / 128, 1), 256, 0, stream>>>(
        a16, HID, W2p, HID, HID, INP(s, 5), xs, HID, nullptr, 0);

    // ---- assemble h, attention scalars, masked softmax ----
    k_build_h<<<(NTOT * HID + 255) / 256, 256, 0, stream>>>(h_in, s, xs, nodeA, hF, hRM, hPK);
    k_srcdst<<<(NTOT + 255) / 256, 256, 0, stream>>>(hF, GRU(s, 0), GRU(s, 1), srcv, dstv);
    k_attn_row<<<NTOT, 256, 0, stream>>>(nodeA, edgeA, srcv, dstv, alphaF, al16);

    // ---- m = alpha @ h (dense WMMA einsum, alpha read exactly once) ----
    k_wmma_gemm<<<dim3(NTOT / 128, 1), 256, 0, stream>>>(
        al16, NTOT, hPK, NTOT, HID, nullptr, mAgg, HID, nullptr, 0);
    k_build_mcat<<<(NTOT * 128 + 255) / 256, 256, 0, stream>>>(hF, mAgg, mcat);
    k_wmma_gemm<<<dim3(NTOT / 128, 1), 256, 0, stream>>>(
        mcat, 128, Wmp, 128, HID, GRU(s, 3), nullptr, 0, m16, HID);

    // ---- fused gate GEMMs + GRU elementwise ----
    k_wmma_gemm<<<dim3(NTOT / 128, 3), 256, 0, stream>>>(
        m16, HID, Wzp, HID, 192, nullptr, GMb, 192, nullptr, 0);
    k_wmma_gemm<<<dim3(NTOT / 128, 3), 256, 0, stream>>>(
        hRM, HID, Uzp, HID, 192, nullptr, GHb, 192, nullptr, 0);
    k_gru<<<(NTOT * HID + 255) / 256, 256, 0, stream>>>(
        GMb, GHb, GRU(s, 6), GRU(s, 9), GRU(s, 12), hF, s, outH);
  }

  k_final<<<(NTOT + 255) / 256, 256, 0, stream>>>(
      outH, WnO, bnO, WeO, beO, nodeA, edgeA, outSig, outY);
}
